// MultiHeadAttention_82386062672073
// MI455X (gfx1250) — compile-verified, matrix operand order checked by reading, not run on hardware
//
#include <hip/hip_runtime.h>
#include <hip/hip_bf16.h>
#include <stdint.h>

// ---------------------------------------------------------------------------
// MI455X (gfx1250) multi-head self-attention, bf16 WMMA with f32 accumulate.
// Reference: q = x@Wq ; Q=K=V=q ; softmax(QK^T/8) @ V.  B=4,S=2048,D=512,H=8.
// ---------------------------------------------------------------------------

#define S_LEN 2048
#define D_DIM 512
#define HEADS 8
#define HD    64

#if defined(__has_builtin)
# if __has_builtin(__builtin_amdgcn_global_load_async_to_lds_b128) && \
     __has_builtin(__builtin_amdgcn_s_wait_asynccnt)
#  define USE_ASYNC_LDS 1
# endif
#endif

typedef __attribute__((ext_vector_type(16))) __bf16 v16bf;
typedef __attribute__((ext_vector_type(8)))  float  v8f;

// Matches the builtin prototype: vector-of-4-int pointee, AS1 (global) / AS3 (LDS)
typedef int v4i_vs __attribute__((vector_size(16)));
typedef __attribute__((address_space(1))) v4i_vs* gptr_b128;
typedef __attribute__((address_space(3))) v4i_vs* lptr_b128;

union FragU {
    v16bf    v;
    uint4    q[2];
    uint16_t s[16];
};

__device__ __forceinline__ uint16_t f32_to_bf16(float f) {
    union { float f; uint32_t u; } c; c.f = f;
    uint32_t u = c.u + 0x7FFFu + ((c.u >> 16) & 1u);   // round-to-nearest-even
    return (uint16_t)(u >> 16);
}

// Load one 16x32 bf16 A/B fragment (wave32 CDNA5 layout).
// Per lane: elements [k0 .. k0+7] and [k0+16 .. k0+23] of row `row`
// (caller passes k0 = chunkBase + (lane>>4)*8, row based on lane&15).
__device__ __forceinline__ v16bf load_frag(const uint16_t* base, int row,
                                           int stride, int k0) {
    FragU f;
    f.q[0] = *(const uint4*)(base + (size_t)row * stride + k0);
    f.q[1] = *(const uint4*)(base + (size_t)row * stride + k0 + 16);
    return f.v;
}

__device__ __forceinline__ void load16f(const float* p, float* v) {
    float4 a = ((const float4*)p)[0];
    float4 b = ((const float4*)p)[1];
    float4 c = ((const float4*)p)[2];
    float4 d = ((const float4*)p)[3];
    v[0]=a.x;  v[1]=a.y;  v[2]=a.z;  v[3]=a.w;
    v[4]=b.x;  v[5]=b.y;  v[6]=b.z;  v[7]=b.w;
    v[8]=c.x;  v[9]=c.y;  v[10]=c.z; v[11]=c.w;
    v[12]=d.x; v[13]=d.y; v[14]=d.z; v[15]=d.w;
}

__device__ __forceinline__ float rowmax16(float v) {
    #pragma unroll
    for (int m = 1; m < 16; m <<= 1) v = fmaxf(v, __shfl_xor(v, m, 32));
    return v;
}
__device__ __forceinline__ float rowsum16(float v) {
    #pragma unroll
    for (int m = 1; m < 16; m <<= 1) v += __shfl_xor(v, m, 32);
    return v;
}

// ---------------------------------------------------------------------------
// Kernel 1: q = x @ Wq   (M=8192, N=512, K=512), bf16 WMMA, bf16 output to ws.
// Block = 256 threads (8 waves). Tile 128x128, K-step 32.
// ---------------------------------------------------------------------------
__global__ __launch_bounds__(256)
void gemm_q_bf16(const float* __restrict__ x, const float* __restrict__ Wq,
                 uint16_t* __restrict__ qbf) {
    __shared__ uint16_t ldsA[128 * 32];   // [m][k]
    __shared__ uint16_t ldsB[128 * 32];   // transposed: [n][k]

    const int t    = threadIdx.x;
    const int wave = t >> 5, lane = t & 31;
    const int half = lane >> 4, l16 = lane & 15;
    const int mBase = blockIdx.x * 128;
    const int nBase = blockIdx.y * 128;
    const int wr = wave & 3;          // 4 M-slabs of 32 rows
    const int wc = wave >> 2;         // 2 N-slabs of 64 cols

    const int arow = t >> 1, acol = (t & 1) * 16;   // A: 128 rows x 32 cols
    const int brow = t >> 3, bcol = (t & 7) * 16;   // B: 32 rows  x 128 cols

    v8f acc[8] = {};   // [mt(2)][nt(4)] 16x16 f32 tiles

    for (int k0 = 0; k0 < D_DIM; k0 += 32) {
        __syncthreads();
        // ---- stage A tile (convert f32 -> bf16) ----
        {
            const float* src = x + (size_t)(mBase + arow) * D_DIM + k0 + acol;
            float v[16]; load16f(src, v);
            FragU f;
            #pragma unroll
            for (int i = 0; i < 16; ++i) f.s[i] = f32_to_bf16(v[i]);
            *(uint4*)&ldsA[arow * 32 + acol]     = f.q[0];
            *(uint4*)&ldsA[arow * 32 + acol + 8] = f.q[1];
        }
        // ---- stage B tile transposed ----
        {
            const float* src = Wq + (size_t)(k0 + brow) * D_DIM + nBase + bcol;
            float v[16]; load16f(src, v);
            #pragma unroll
            for (int i = 0; i < 16; ++i)
                ldsB[(bcol + i) * 32 + brow] = f32_to_bf16(v[i]);
        }
        // ---- prefetch next K-step tiles into cache (global_prefetch_b8) ----
        if (k0 + 32 < D_DIM) {
            __builtin_prefetch(x + (size_t)(mBase + arow) * D_DIM + k0 + 32 + acol, 0, 1);
            __builtin_prefetch(Wq + (size_t)(k0 + 32 + brow) * D_DIM + nBase + bcol, 0, 1);
        }
        __syncthreads();

        // ---- WMMA compute ----
        v16bf af[2], bf[4];
        #pragma unroll
        for (int mt = 0; mt < 2; ++mt)
            af[mt] = load_frag(ldsA, wr * 32 + mt * 16 + l16, 32, half * 8);
        #pragma unroll
        for (int nt = 0; nt < 4; ++nt)
            bf[nt] = load_frag(ldsB, wc * 64 + nt * 16 + l16, 32, half * 8);
        #pragma unroll
        for (int mt = 0; mt < 2; ++mt)
            #pragma unroll
            for (int nt = 0; nt < 4; ++nt)
                acc[mt * 4 + nt] = __builtin_amdgcn_wmma_f32_16x16x32_bf16(
                    false, af[mt], false, bf[nt], (short)0, acc[mt * 4 + nt],
                    false, false);
    }

    // ---- store q as bf16 (C/D layout: element r -> row r + 8*half) ----
    #pragma unroll
    for (int mt = 0; mt < 2; ++mt)
        #pragma unroll
        for (int nt = 0; nt < 4; ++nt)
            #pragma unroll
            for (int r = 0; r < 8; ++r) {
                int row = mBase + wr * 32 + mt * 16 + r + 8 * half;
                int col = nBase + wc * 64 + nt * 16 + l16;
                qbf[(size_t)row * D_DIM + col] = f32_to_bf16(acc[mt * 4 + nt][r]);
            }
}

// ---------------------------------------------------------------------------
// Kernel 2: flash attention per (128-query tile, b, h).  Q=K=V=q (bf16).
// Block = 256 threads (8 waves); each wave owns 16 query rows, full hd=64.
// Key blocks of 64: staged via GLOBAL_LOAD_ASYNC_TO_LDS (ASYNCcnt) into
// [key][hd], then transposed LDS->LDS into [hd][key] for V fragments.
// ---------------------------------------------------------------------------
__global__ __launch_bounds__(256)
void attn_flash(const uint16_t* __restrict__ qbf, float* __restrict__ out) {
    __shared__ uint16_t kv_khd[64 * 64];        // [key][hd]  -> K^T fragments
    __shared__ uint16_t kv_hdk[64 * 64];        // [hd][key]  -> V fragments
    __shared__ uint16_t pslab[8 * 16 * 64];     // per-wave P (D-layout -> A-layout)

    const int t    = threadIdx.x;
    const int wave = t >> 5, lane = t & 31;
    const int half = lane >> 4, l16 = lane & 15;
    const int bh = blockIdx.y;
    const int b = bh >> 3, h = bh & 7;
    const int qr = blockIdx.x * 128 + wave * 16;          // first query row of wave
    const size_t headBase = (size_t)b * S_LEN * D_DIM + (size_t)h * HD;

    // Q fragments for this wave's 16 rows, straight from global (contiguous 16B)
    v16bf qf[2];
    #pragma unroll
    for (int kk = 0; kk < 2; ++kk)
        qf[kk] = load_frag(qbf + headBase + (size_t)qr * D_DIM, l16, D_DIM,
                           kk * 32 + half * 8);

    v8f accO[4] = {};                  // 16x64 output accumulator (4 hd tiles)
    float mrow[8], lrow[8];
    #pragma unroll
    for (int r = 0; r < 8; ++r) { mrow[r] = -__builtin_inff(); lrow[r] = 0.0f; }

    const int keyr = t >> 2, hdB = (t & 3) * 16;   // coop-load mapping

    for (int j = 0; j < S_LEN / 64; ++j) {
        const int key0 = j * 64;
        const uint16_t* gsrc =
            qbf + headBase + (size_t)(key0 + keyr) * D_DIM + hdB;
        __syncthreads();

        // ---- stage KV block [key][hd] ----
#ifdef USE_ASYNC_LDS
        __builtin_amdgcn_global_load_async_to_lds_b128(
            (gptr_b128)(void*)gsrc,
            (lptr_b128)(void*)&kv_khd[keyr * 64 + hdB], 0, 0);
        __builtin_amdgcn_global_load_async_to_lds_b128(
            (gptr_b128)(void*)(gsrc + 8),
            (lptr_b128)(void*)&kv_khd[keyr * 64 + hdB + 8], 0, 0);
        __builtin_amdgcn_s_wait_asynccnt(0);
        __syncthreads();
        // ---- build transposed [hd][key] copy, LDS -> regs -> LDS ----
        {
            FragU f;
            f.q[0] = *(const uint4*)&kv_khd[keyr * 64 + hdB];
            f.q[1] = *(const uint4*)&kv_khd[keyr * 64 + hdB + 8];
            #pragma unroll
            for (int i = 0; i < 16; ++i)
                kv_hdk[(hdB + i) * 64 + keyr] = f.s[i];
        }
#else
        {
            FragU f;
            f.q[0] = *(const uint4*)(gsrc);
            f.q[1] = *(const uint4*)(gsrc + 8);
            *(uint4*)&kv_khd[keyr * 64 + hdB]     = f.q[0];
            *(uint4*)&kv_khd[keyr * 64 + hdB + 8] = f.q[1];
            #pragma unroll
            for (int i = 0; i < 16; ++i)
                kv_hdk[(hdB + i) * 64 + keyr] = f.s[i];
        }
#endif
        // ---- prefetch next KV block toward L2/L0 ----
        if (j + 1 < S_LEN / 64)
            __builtin_prefetch(gsrc + 64 * D_DIM, 0, 1);
        __syncthreads();

        // ---- scores: S = Q (16x64) @ K^T (64x64), scaled by 1/sqrt(64) ----
        v8f sc[4];
        #pragma unroll
        for (int nt = 0; nt < 4; ++nt) {
            v8f z = {};
            v16bf k0f = load_frag(kv_khd, nt * 16 + l16, 64, 0 * 32 + half * 8);
            v16bf k1f = load_frag(kv_khd, nt * 16 + l16, 64, 1 * 32 + half * 8);
            z = __builtin_amdgcn_wmma_f32_16x16x32_bf16(false, qf[0], false, k0f,
                                                        (short)0, z, false, false);
            z = __builtin_amdgcn_wmma_f32_16x16x32_bf16(false, qf[1], false, k1f,
                                                        (short)0, z, false, false);
            sc[nt] = z * 0.125f;
        }

        // ---- online softmax (row = r + 8*half, across 16-lane groups) ----
        float alpha[8];
        #pragma unroll
        for (int r = 0; r < 8; ++r) {
            float bm = sc[0][r];
            #pragma unroll
            for (int nt = 1; nt < 4; ++nt) bm = fmaxf(bm, sc[nt][r]);
            bm = rowmax16(bm);
            float mnew = fmaxf(mrow[r], bm);
            alpha[r] = __expf(mrow[r] - mnew);
            mrow[r] = mnew;
        }
        float rs[8];
        #pragma unroll
        for (int r = 0; r < 8; ++r) rs[r] = 0.0f;
        #pragma unroll
        for (int nt = 0; nt < 4; ++nt)
            #pragma unroll
            for (int r = 0; r < 8; ++r) {
                float p = __expf(sc[nt][r] - mrow[r]);
                sc[nt][r] = p;
                rs[r] += p;
            }
        #pragma unroll
        for (int r = 0; r < 8; ++r) {
            lrow[r] = lrow[r] * alpha[r] + rowsum16(rs[r]);
            #pragma unroll
            for (int nt = 0; nt < 4; ++nt) accO[nt][r] *= alpha[r];
        }

        // ---- P: D-layout regs -> per-wave LDS slab (bf16, A-layout readable) --
        uint16_t* ps = &pslab[wave * 16 * 64];
        #pragma unroll
        for (int nt = 0; nt < 4; ++nt)
            #pragma unroll
            for (int r = 0; r < 8; ++r)
                ps[(r + 8 * half) * 64 + nt * 16 + l16] = f32_to_bf16(sc[nt][r]);
        __syncthreads();

        v16bf pf0 = load_frag(ps, l16, 64, 0 * 32 + half * 8);
        v16bf pf1 = load_frag(ps, l16, 64, 1 * 32 + half * 8);

        // ---- accO += P (16x64) @ V (64x64) ----
        #pragma unroll
        for (int nt = 0; nt < 4; ++nt) {
            v16bf v0f = load_frag(kv_hdk, nt * 16 + l16, 64, 0 * 32 + half * 8);
            v16bf v1f = load_frag(kv_hdk, nt * 16 + l16, 64, 1 * 32 + half * 8);
            accO[nt] = __builtin_amdgcn_wmma_f32_16x16x32_bf16(
                false, pf0, false, v0f, (short)0, accO[nt], false, false);
            accO[nt] = __builtin_amdgcn_wmma_f32_16x16x32_bf16(
                false, pf1, false, v1f, (short)0, accO[nt], false, false);
        }
    }

    // ---- epilogue: normalize and store f32 output ----
    float inv[8];
    #pragma unroll
    for (int r = 0; r < 8; ++r) inv[r] = 1.0f / lrow[r];
    #pragma unroll
    for (int nt = 0; nt < 4; ++nt)
        #pragma unroll
        for (int r = 0; r < 8; ++r) {
            int row = qr + r + 8 * half;
            out[headBase + (size_t)row * D_DIM + nt * 16 + l16] =
                accO[nt][r] * inv[r];
        }
}

// ---------------------------------------------------------------------------
extern "C" void kernel_launch(void* const* d_in, const int* in_sizes, int n_in,
                              void* d_out, int out_size, void* d_ws, size_t ws_size,
                              hipStream_t stream) {
    const float* x  = (const float*)d_in[0];   // [4,2048,512] f32
    const float* Wq = (const float*)d_in[1];   // [512,512]    f32
    float* out      = (float*)d_out;           // [4,2048,512] f32
    uint16_t* qbf   = (uint16_t*)d_ws;         // q in bf16, 8 MB scratch

    dim3 gGemm(8192 / 128, 512 / 128);         // (64, 4)
    gemm_q_bf16<<<gGemm, 256, 0, stream>>>(x, Wq, qbf);

    dim3 gAttn(S_LEN / 128, 4 * HEADS);        // (16, 32)
    attn_flash<<<gAttn, 256, 0, stream>>>(qbf, out);
}
